// BigramModel_33002528703064
// MI455X (gfx1250) — matrix operand order repaired and checked by reading, not compiled, not run on hardware
//
#include <hip/hip_runtime.h>
#include <hip/hip_bf16.h>

typedef _Float16 h16;
typedef __attribute__((ext_vector_type(8)))  _Float16 v8h;
typedef __attribute__((ext_vector_type(16))) _Float16 v16h;
typedef __attribute__((ext_vector_type(8)))  float    v8f;
typedef __attribute__((ext_vector_type(4)))  unsigned int u32x4;
typedef __attribute__((ext_vector_type(4)))  int      i32x4;
typedef __attribute__((ext_vector_type(8)))  int      i32x8;

#define BB   4096
#define TT   128
#define DD   64
#define VV   65
#define VP   80      // padded vocab (5 N-tiles of 16)
#define LL   3
#define HH   8
#define HDD  8
#define DFF  256

// f16 workspace layout (offsets in halves); all matrices stored pre-swizzled
// into the WMMA B-operand fragment layout:
//   swz[((kt*NT + nt)*32 + lane)*16 + e] = W[kt*32 + ((lane&16)?16:0) + e][nt*16 + (lane&15)]
#define OFF_WQ 0
#define OFF_WK 12288
#define OFF_WV 24576
#define OFF_WO 36864
#define OFF_W1 49152
#define OFF_W2 98304
#define OFF_LM 147456
#define WS_HALVES 152576

#ifndef __has_builtin
#define __has_builtin(x) 0
#endif
#if __has_builtin(__builtin_amdgcn_tensor_load_to_lds)
#define HAVE_TDM 1
#else
#define HAVE_TDM 0
#endif

#if __has_builtin(__builtin_amdgcn_s_wait_tensorcnt)
#define WAIT_TCNT() __builtin_amdgcn_s_wait_tensorcnt(0)
#else
#define WAIT_TCNT() asm volatile("s_wait_tensorcnt 0x0" ::: "memory")
#endif

// ---------------------------------------------------------------------------
// Weight conversion fp32 -> f16 swizzled B-fragments + zero loss slot
// ---------------------------------------------------------------------------
__device__ __forceinline__ h16 swz_fetch(const float* src, int i2, int NT,
                                         int NvalidCols) {
  int tile = i2 >> 9;          // / 512 halves per 32x16 tile
  int lane = (i2 >> 4) & 31;
  int e    = i2 & 15;
  int kt = tile / NT, nt = tile % NT;
  int k = kt * 32 + ((lane & 16) ? 16 : 0) + e;
  int n = nt * 16 + (lane & 15);
  return (n < NvalidCols) ? (h16)src[k * NvalidCols + n] : (h16)0.f;
}

__global__ void convert_weights(const float* wq, const float* wk, const float* wv,
                                const float* wo, const float* w1, const float* w2,
                                const float* lm_w, h16* ws, float* loss_slot) {
  int i = blockIdx.x * blockDim.x + threadIdx.x;
  if (i == 0) *loss_slot = 0.f;
  if (i >= WS_HALVES) return;
  if (i < OFF_W1) {                           // wq|wk|wv|wo, each [L][64x64]
    int grp = i / 12288, r = i % 12288, l = r / 4096, i2 = r % 4096;
    const float* src = ((grp == 0) ? wq : (grp == 1) ? wk : (grp == 2) ? wv : wo)
                       + l * DD * DD;
    ws[i] = swz_fetch(src, i2, 4, DD);
  } else if (i < OFF_W2) {                    // w1 [L][64x256]
    int j = i - OFF_W1, l = j / 16384, i2 = j % 16384;
    ws[i] = swz_fetch(w1 + l * DD * DFF, i2, 16, DFF);
  } else if (i < OFF_LM) {                    // w2 [L][256x64]
    int j = i - OFF_W2, l = j / 16384, i2 = j % 16384;
    ws[i] = swz_fetch(w2 + l * DFF * DD, i2, 4, DD);
  } else {                                    // lm_w [64x65] padded to 64x80
    ws[i] = swz_fetch(lm_w, i - OFF_LM, 5, VV);
  }
}

// ---------------------------------------------------------------------------
// One 16x16 output tile of C = A(LDS f16, row-major) @ B(pre-swizzled frags)
// ---------------------------------------------------------------------------
__device__ __forceinline__ v8f gemm_tile(const h16* aL, int lda,
                                         const h16* bSwz, int NT,
                                         int Ktot, int mt, int nt) {
  v8f c = {};
  const int lane = threadIdx.x & 31;
  const int m   = mt * 16 + (lane & 15);
  const int ka0 = (lane & 16) ? 16 : 0;
  const h16* arow = aL + m * lda + ka0;
#pragma unroll
  for (int k0 = 0; k0 < Ktot; k0 += 32) {
    v8h alo = *(const v8h*)(arow + k0);
    v8h ahi = *(const v8h*)(arow + k0 + 8);
    v16h a = __builtin_shufflevector(alo, ahi, 0, 1, 2, 3, 4, 5, 6, 7,
                                     8, 9, 10, 11, 12, 13, 14, 15);
    v16h b = *(const v16h*)(bSwz + (((k0 >> 5) * NT + nt) * 32 + lane) * 16);
    c = __builtin_amdgcn_wmma_f32_16x16x32_f16(false, a, false, b, (short)0, c,
                                               false, false);
  }
  return c;
}

__device__ __forceinline__ void ln_rows(const float* x, const float* g,
                                        const float* be, h16* out, int tid) {
  if (tid < TT) {
    const float* row = x + tid * DD;
    float mu = 0.f;
    for (int d = 0; d < DD; ++d) mu += row[d];
    mu *= (1.f / DD);
    float var = 0.f;
    for (int d = 0; d < DD; ++d) { float t = row[d] - mu; var += t * t; }
    float rs = rsqrtf(var * (1.f / DD) + 1e-5f);
    for (int d = 0; d < DD; ++d)
      out[tid * DD + d] = (h16)((row[d] - mu) * rs * g[d] + be[d]);
  }
}

// ---------------------------------------------------------------------------
// Full forward: one block per sequence, 8 wave32s
// ---------------------------------------------------------------------------
__global__ __launch_bounds__(256) void gpt_forward(
    const float* tok_emb, const float* pos_emb,
    const float* ln1_g, const float* ln1_b, const float* bo,
    const float* ln2_g, const float* ln2_b,
    const float* b1, const float* b2,
    const float* lnf_g, const float* lnf_b, const float* lm_b,
    const int* idx, const int* targets,
    const h16* ws, float* out_logits, float* out_loss) {
  __shared__ __align__(64) unsigned char pool[163840];
  float* sX    = (float*)pool;                 // [128][64] fp32 residual
  h16*   sXN   = (h16*)(pool + 32768);         // [128][64] f16 LN/attn-out
  h16*   sQ    = (h16*)(pool + 49152);         // [128][64]
  h16*   sK    = (h16*)(pool + 65536);         // [128][64]
  h16*   sVT   = (h16*)(pool + 81920);         // [64][128] V transposed
  h16*   sSc   = (h16*)(pool + 98304);         // [8][16][128] scores/probs
  h16*   sH1   = (h16*)(pool + 49152);         // [128][256] (reuses Q/K/VT+Sc)
  float* sLg   = (float*)(pool + 49152);       // [128][80]  (reuses Q/K/VT)
  float* sRed  = (float*)(pool + 98304);       // [256]      (reuses Sc)
  h16*   sW1   = (h16*)(pool + 131072);        // [16384] TDM-staged w1 block

  const int b    = blockIdx.x;
  const int tid  = threadIdx.x;
  const int wave = tid >> 5;
  const int lane = tid & 31;

  // warm L2/WGP$ for this block's weight stream (global_prefetch_b8)
  if (lane == 0)
    __builtin_prefetch(ws + (size_t)wave * (WS_HALVES / 8), 0, 1);

  // embeddings
  for (int i = tid; i < TT * DD; i += 256) {
    int t = i >> 6, d = i & 63;
    sX[i] = tok_emb[idx[b * TT + t] * DD + d] + pos_emb[i];
  }
  __syncthreads();

  for (int l = 0; l < LL; ++l) {
#if HAVE_TDM
    // Kick the layer's w1 fragment block (32KB, linear) LDS-ward via the
    // Tensor Data Mover; it lands while QKV + attention execute.
    if (wave == 0) {
      unsigned int lds_off = (unsigned int)(uintptr_t)(void*)sW1;
      unsigned long long ga =
          (unsigned long long)(uintptr_t)(ws + OFF_W1 + l * DD * DFF);
      u32x4 g0 = {1u,                                   // count=1 valid D#
                  lds_off,                              // lds_addr
                  (unsigned int)ga,                     // global_addr[31:0]
                  (unsigned int)(((ga >> 32) & 0x1FFFFFFu) | (2u << 30))};
      i32x8 g1 = {0x00010000,      // data_size=2B
                  0x40000000,      // tensor_dim0 lo16=16384 in [31:16]
                  0x00010000,      // dim0 hi=0 | tensor_dim1 lo16=1
                  0x40000000,      // dim1 hi=0 | tile_dim0=16384
                  1,               // tile_dim1=1, tile_dim2=0
                  16384,           // tensor_dim0_stride lo32
                  0, 0};
      i32x4 gz4 = {0, 0, 0, 0};
#if __clang_major__ >= 23
      i32x8 gz8 = {0, 0, 0, 0, 0, 0, 0, 0};
      __builtin_amdgcn_tensor_load_to_lds(g0, g1, gz4, gz4, gz8, 0);
#else
      __builtin_amdgcn_tensor_load_to_lds(g0, g1, gz4, gz4, 0);
#endif
    }
#endif
    ln_rows(sX, ln1_g + l * DD, ln1_b + l * DD, sXN, tid);
    __syncthreads();

    // ---- QKV: 3 x [128x64]@[64x64]; uniform loops per matrix ----
    const h16* Wq = ws + OFF_WQ + l * DD * DD;
    const h16* Wk = ws + OFF_WK + l * DD * DD;
    const h16* Wv = ws + OFF_WV + l * DD * DD;
#pragma unroll
    for (int t4 = 0; t4 < 4; ++t4) {        // Q
      int tile = wave + t4 * 8, mt = tile >> 2, nt = tile & 3;
      v8f c = gemm_tile(sXN, DD, Wq, 4, DD, mt, nt);
      int n = nt * 16 + (lane & 15), rb = mt * 16 + ((lane & 16) ? 8 : 0);
#pragma unroll
      for (int r = 0; r < 8; ++r) sQ[(rb + r) * DD + n] = (h16)c[r];
    }
#pragma unroll
    for (int t4 = 0; t4 < 4; ++t4) {        // K
      int tile = wave + t4 * 8, mt = tile >> 2, nt = tile & 3;
      v8f c = gemm_tile(sXN, DD, Wk, 4, DD, mt, nt);
      int n = nt * 16 + (lane & 15), rb = mt * 16 + ((lane & 16) ? 8 : 0);
#pragma unroll
      for (int r = 0; r < 8; ++r) sK[(rb + r) * DD + n] = (h16)c[r];
    }
#pragma unroll
    for (int t4 = 0; t4 < 4; ++t4) {        // V, stored transposed
      int tile = wave + t4 * 8, mt = tile >> 2, nt = tile & 3;
      v8f c = gemm_tile(sXN, DD, Wv, 4, DD, mt, nt);
      int n = nt * 16 + (lane & 15), rb = mt * 16 + ((lane & 16) ? 8 : 0);
      v8h vv;
#pragma unroll
      for (int r = 0; r < 8; ++r) vv[r] = (h16)c[r];
      *(v8h*)(sVT + n * TT + rb) = vv;
    }
    __syncthreads();

    // ---- attention: wave == head ----
    {
      const int h = wave;
      const float scale = 0.35355339059f; // 1/sqrt(8)
      h16* sc = sSc + h * 16 * 128;
      const v8h z8 = {};
      for (int mt = 0; mt < 8; ++mt) {
        // A fragment (q rows, K=8 zero-padded to 32) is constant over kt
        int m = mt * 16 + (lane & 15);
        v8h qv = (lane < 16) ? *(const v8h*)(sQ + m * DD + h * HDD) : z8;
        v16h a = __builtin_shufflevector(qv, z8, 0, 1, 2, 3, 4, 5, 6, 7,
                                         8, 9, 10, 11, 12, 13, 14, 15);
#pragma unroll
        for (int kt = 0; kt < 8; ++kt) {
          int key = kt * 16 + (lane & 15);
          v8h kv = (lane < 16) ? *(const v8h*)(sK + key * DD + h * HDD) : z8;
          v16h bfr = __builtin_shufflevector(kv, z8, 0, 1, 2, 3, 4, 5, 6, 7,
                                             8, 9, 10, 11, 12, 13, 14, 15);
          v8f c = {};
          c = __builtin_amdgcn_wmma_f32_16x16x32_f16(false, a, false, bfr,
                                                     (short)0, c, false, false);
          int rb  = (lane & 16) ? 8 : 0;
          int col = kt * 16 + (lane & 15);
#pragma unroll
          for (int r = 0; r < 8; ++r) {
            int row = mt * 16 + rb + r;
            float s = (col <= row) ? c[r] * scale : -60000.f;
            sc[(rb + r) * 128 + col] = (h16)s;
          }
        }
        __builtin_amdgcn_wave_barrier();
        asm volatile("s_wait_dscnt 0" ::: "memory");
        // softmax over the 16 rows (lanes 0..15, one row each)
        if (lane < 16) {
          h16* srow = sc + lane * 128;
          float mx = -1e30f;
          for (int j = 0; j < TT; ++j) mx = fmaxf(mx, (float)srow[j]);
          float sum = 0.f;
          for (int j = 0; j < TT; ++j) {
            float e = __expf((float)srow[j] - mx);
            sum += e;
            srow[j] = (h16)e;
          }
          float inv = 1.f / sum;
          for (int j = 0; j < TT; ++j) srow[j] = (h16)((float)srow[j] * inv);
        }
        __builtin_amdgcn_wave_barrier();
        asm volatile("s_wait_dscnt 0" ::: "memory");
        // probs[16x128] @ v[128x8] (N padded 8->16), V read transposed
        {
          v8f c = {};
          int n   = lane & 15;
          int ka0 = (lane & 16) ? 16 : 0;
          int kb0 = (lane & 16) ? 16 : 0;
          const h16* prow = sc + (lane & 15) * 128 + ka0;
          const h16* vrow = sVT + (h * HDD + n) * TT + kb0;  // n<8 only
          const v16h z16 = {};
#pragma unroll
          for (int k0 = 0; k0 < TT; k0 += 32) {
            v8h alo = *(const v8h*)(prow + k0);
            v8h ahi = *(const v8h*)(prow + k0 + 8);
            v16h a2 = __builtin_shufflevector(alo, ahi, 0, 1, 2, 3, 4, 5, 6, 7,
                                              8, 9, 10, 11, 12, 13, 14, 15);
            v16h b2 = (n < HDD) ? *(const v16h*)(vrow + k0) : z16;
            c = __builtin_amdgcn_wmma_f32_16x16x32_f16(false, a2, false, b2,
                                                       (short)0, c, false, false);
          }
          if (n < HDD) {
            int rb = mt * 16 + ((lane & 16) ? 8 : 0);
#pragma unroll
            for (int r = 0; r < 8; ++r)
              sXN[(rb + r) * DD + h * HDD + n] = (h16)c[r];
          }
        }
      }
    }
    __syncthreads();

    // ---- o-proj + bias + residual ----
#pragma unroll
    for (int t4 = 0; t4 < 4; ++t4) {
      int tile = wave + t4 * 8, mt = tile >> 2, nt = tile & 3;
      v8f c = gemm_tile(sXN, DD, ws + OFF_WO + l * DD * DD, 4, DD, mt, nt);
      int n = nt * 16 + (lane & 15), rb = mt * 16 + ((lane & 16) ? 8 : 0);
      float bias = bo[l * DD + n];
#pragma unroll
      for (int r = 0; r < 8; ++r) sX[(rb + r) * DD + n] += c[r] + bias;
    }
    __syncthreads();

    ln_rows(sX, ln2_g + l * DD, ln2_b + l * DD, sXN, tid);
#if HAVE_TDM
    if (wave == 0) WAIT_TCNT();           // w1 fragments resident in LDS
#else
    for (int i = tid * 8; i < DD * DFF; i += 256 * 8)
      *(v8h*)(sW1 + i) = *(const v8h*)(ws + OFF_W1 + l * DD * DFF + i);
#endif
    __syncthreads();

    // ---- MLP fc1 + relu (B fragments from TDM-staged LDS) ----
    for (int tile = wave; tile < 128; tile += 8) {
      int mt = tile >> 4, nt = tile & 15;
      v8f c = gemm_tile(sXN, DD, sW1, 16, DD, mt, nt);
      int n = nt * 16 + (lane & 15), rb = mt * 16 + ((lane & 16) ? 8 : 0);
      float bias = b1[l * DFF + n];
#pragma unroll
      for (int r = 0; r < 8; ++r)
        sH1[(rb + r) * DFF + n] = (h16)fmaxf(c[r] + bias, 0.f);
    }
    __syncthreads();

    // ---- MLP fc2 + bias + residual ----
#pragma unroll
    for (int t4 = 0; t4 < 4; ++t4) {
      int tile = wave + t4 * 8, mt = tile >> 2, nt = tile & 3;
      v8f c = gemm_tile(sH1, DFF, ws + OFF_W2 + l * DFF * DD, 4, DFF, mt, nt);
      int n = nt * 16 + (lane & 15), rb = mt * 16 + ((lane & 16) ? 8 : 0);
      float bias = b2[l * DD + n];
#pragma unroll
      for (int r = 0; r < 8; ++r) sX[(rb + r) * DD + n] += c[r] + bias;
    }
    __syncthreads();
  }

  // ---- final LN + LM head ----
  ln_rows(sX, lnf_g, lnf_b, sXN, tid);
  __syncthreads();

  for (int tile = wave; tile < 40; tile += 8) {
    int mt = tile / 5, nt = tile % 5;
    v8f c = gemm_tile(sXN, DD, ws + OFF_LM, 5, DD, mt, nt);
    int n = nt * 16 + (lane & 15), rb = mt * 16 + ((lane & 16) ? 8 : 0);
    float bias = (n < VV) ? lm_b[n] : 0.f;
#pragma unroll
    for (int r = 0; r < 8; ++r) {
      int row = rb + r;
      float v = (n < VV) ? (c[r] + bias) : -60000.f;
      sLg[row * VP + n] = v;
      if (n < VV) out_logits[((size_t)(b * TT + row)) * VV + n] = v;
    }
  }
  __syncthreads();

  // ---- loss: per-row logsumexp, block reduce, one atomic per block ----
  float part = 0.f;
  if (tid < TT) {
    const float* lr = sLg + tid * VP;
    float mx = -1e30f;
    for (int j = 0; j < VV; ++j) mx = fmaxf(mx, lr[j]);
    float s = 0.f;
    for (int j = 0; j < VV; ++j) s += __expf(lr[j] - mx);
    int tgt = targets[b * TT + tid];
    part = -(lr[tgt] - mx - __logf(s));
  }
  sRed[tid] = part;
  __syncthreads();
  if (tid == 0) {
    float s = 0.f;
    for (int i = 0; i < 256; ++i) s += sRed[i];
    atomicAdd(out_loss, s * (1.f / ((float)BB * TT)));
  }
}

// ---------------------------------------------------------------------------
extern "C" void kernel_launch(void* const* d_in, const int* in_sizes, int n_in,
                              void* d_out, int out_size, void* d_ws, size_t ws_size,
                              hipStream_t stream) {
  const float* tok_emb = (const float*)d_in[0];
  const float* pos_emb = (const float*)d_in[1];
  const float* ln1_g   = (const float*)d_in[2];
  const float* ln1_b   = (const float*)d_in[3];
  const float* wq      = (const float*)d_in[4];
  const float* wk      = (const float*)d_in[5];
  const float* wv      = (const float*)d_in[6];
  const float* wo      = (const float*)d_in[7];
  const float* bo      = (const float*)d_in[8];
  const float* ln2_g   = (const float*)d_in[9];
  const float* ln2_b   = (const float*)d_in[10];
  const float* w1      = (const float*)d_in[11];
  const float* b1      = (const float*)d_in[12];
  const float* w2      = (const float*)d_in[13];
  const float* b2      = (const float*)d_in[14];
  const float* lnf_g   = (const float*)d_in[15];
  const float* lnf_b   = (const float*)d_in[16];
  const float* lm_w    = (const float*)d_in[17];
  const float* lm_b    = (const float*)d_in[18];
  const int*   idx     = (const int*)d_in[19];
  const int*   targets = (const int*)d_in[20];

  float* out = (float*)d_out;
  h16*   ws  = (h16*)d_ws;
  float* loss_slot = out + (size_t)BB * TT * VV;

  convert_weights<<<(WS_HALVES + 255) / 256, 256, 0, stream>>>(
      wq, wk, wv, wo, w1, w2, lm_w, ws, loss_slot);
  gpt_forward<<<BB, 256, 0, stream>>>(
      tok_emb, pos_emb, ln1_g, ln1_b, bo, ln2_g, ln2_b, b1, b2,
      lnf_g, lnf_b, lm_b, idx, targets, ws, out, loss_slot);
}
